// dot_attention_23854248362444
// MI455X (gfx1250) — matrix operand order repaired and checked by reading, not compiled, not run on hardware
//
#include <hip/hip_runtime.h>
#include <hip/hip_bf16.h>
#include <math.h>

// Problem constants (match reference)
#define BB   8
#define TXN  256
#define TYN  256
#define HH   128
#define CC   256      // 2*H feature dim
#define TG   4        // ty values processed per workgroup
#define LDX  264      // padded LDS row stride in halfs (528B: 16B-aligned, bank-skewed)
#define LDW  264

typedef __attribute__((ext_vector_type(2)))  __bf16 bf16v2;
typedef __attribute__((ext_vector_type(8)))  __bf16 bf16x8;
typedef __attribute__((ext_vector_type(16))) __bf16 bf16x16;
typedef __attribute__((ext_vector_type(8)))  float  f32x8;
typedef __attribute__((ext_vector_type(2)))  float  f32x2;

__device__ __forceinline__ bf16x16 cat16(bf16x8 lo, bf16x8 hi) {
  return __builtin_shufflevector(lo, hi, 0,1,2,3,4,5,6,7,8,9,10,11,12,13,14,15);
}

// Branch-free tanh on the CDNA5 V_TANH_F32 transcendental unit.
__device__ __forceinline__ float fast_tanh(float v) {
#if __has_builtin(__builtin_amdgcn_tanhf)
  return __builtin_amdgcn_tanhf(v);
#else
  float r;
  // v_nop satisfies the TRANS-op hazard (1 independent op before result use).
  asm("v_tanh_f32 %0, %1\n\tv_nop" : "=v"(r) : "v"(v));
  return r;
#endif
}

__launch_bounds__(256, 1)
__global__ void dot_attention_kernel(const float* __restrict__ x,
                                     const float* __restrict__ y,
                                     const float* __restrict__ Wd,
                                     const float* __restrict__ vd,
                                     float* __restrict__ out)
{
  __shared__ __bf16 Xs [TXN * LDX];   // x rows of this batch, bf16
  __shared__ __bf16 Wys[HH  * LDW];   // Wy = Wd * y_row, bf16 (rebuilt per ty)
  __shared__ float  s_lds[TXN];       // scores -> exp weights
  __shared__ float  vd_s[HH];
  __shared__ float  red_s[8];

  const int tid   = threadIdx.x;
  const int lane  = tid & 31;
  const int wave  = tid >> 5;
  const int l15   = lane & 15;        // N index inside a 16x16 tile / row-in-tile
  const int lhalf = lane >> 4;        // which K-half this lane holds
  const int b     = blockIdx.x / (TYN / TG);
  const int ty0   = (blockIdx.x % (TYN / TG)) * TG;

  // ---------------- stage X (f32 -> bf16) and vd into LDS ----------------
  const float* xb = x + (size_t)b * TXN * CC;
  for (int i = tid; i < TXN * CC; i += 256) {
    Xs[(i >> 8) * LDX + (i & 255)] = (__bf16)xb[i];
  }
  if (tid < HH) vd_s[tid] = vd[tid];
  __syncthreads();

  // ------------- persistent A fragments: this wave's 32 tx rows -------------
  // 16-bit A 16x32 layout: lanes 0-15 hold K 0-7 & 16-23, lanes 16-31 hold 8-15 & 24-31.
  const int mrow0 = wave * 32;
  bf16x16 afrag[2][8];
  #pragma unroll
  for (int mt = 0; mt < 2; ++mt) {
    const int row = mrow0 + mt * 16 + l15;
    #pragma unroll
    for (int k = 0; k < 8; ++k) {
      const int cb = k * 32 + lhalf * 8;
      bf16x8 lo = *(const bf16x8*)&Xs[row * LDX + cb];
      bf16x8 hi = *(const bf16x8*)&Xs[row * LDX + cb + 16];
      afrag[mt][k] = cat16(lo, hi);
    }
  }

  // Each thread owns a fixed column pair (c, c+1) of the Wy build.
  const int cpr = (tid & 127) * 2;

  for (int tyi = 0; tyi < TG; ++tyi) {
    const int ty = ty0 + tyi;
    __syncthreads();  // previous iteration's s_lds readers done before rebuild

    // ------------- build Wy[h][c] = Wd[h][c] * y[b,ty,c] (bf16, packed) -------------
    {
      const float* yrow = y + ((size_t)b * TYN + ty) * CC;
      const float y0 = yrow[cpr];
      const float y1 = yrow[cpr + 1];
      // idx walks h with stride 2, h parity = tid>>7; one ds_store_b32 per row.
      for (int idx = tid; idx < HH * (CC / 2); idx += 256) {
        const int h = idx >> 7;
        const f32x2 w = *(const f32x2*)&Wd[h * CC + cpr];   // global_load_b64
        bf16v2 pk;
        pk.x = (__bf16)(w.x * y0);
        pk.y = (__bf16)(w.y * y1);
        *(bf16v2*)&Wys[h * LDW + cpr] = pk;
      }
    }
    __syncthreads();

    // ---- S-rows for this wave: 2 Mtiles x 8 Ntiles x 8 Ksteps of WMMA ----
    // B layout: N = lane%16; lanes 0-15 hold K 0-7 & 16-23, lanes 16-31 hold 8-15 & 24-31.
    const int hrow = l15;  // within-tile N; tile selected by nt
    float p[2][8];
    #pragma unroll
    for (int mt = 0; mt < 2; ++mt)
      #pragma unroll
      for (int r = 0; r < 8; ++r) p[mt][r] = 0.f;

    #pragma unroll
    for (int nt = 0; nt < 8; ++nt) {
      const __bf16* brow = &Wys[(nt * 16 + hrow) * LDW + lhalf * 8];
      f32x8 acc0 = {}, acc1 = {};

      // one-deep software pipeline: issue k+1's B loads before k's WMMAs
      bf16x16 bcur = cat16(*(const bf16x8*)&brow[0],
                           *(const bf16x8*)&brow[16]);
      #pragma unroll
      for (int k = 0; k < 8; ++k) {
        bf16x16 bnext;
        if (k < 7) {
          const int cb = (k + 1) * 32;
          bnext = cat16(*(const bf16x8*)&brow[cb],
                        *(const bf16x8*)&brow[cb + 16]);
        }
        acc0 = __builtin_amdgcn_wmma_f32_16x16x32_bf16(
                 false, afrag[0][k], false, bcur, (short)0, acc0, false, false);
        acc1 = __builtin_amdgcn_wmma_f32_16x16x32_bf16(
                 false, afrag[1][k], false, bcur, (short)0, acc1, false, false);
        bcur = bnext;
      }

      // fuse tanh + vd-dot straight out of the accumulators (branch-free)
      const float vdn = vd_s[nt * 16 + l15];
      #pragma unroll
      for (int r = 0; r < 8; ++r) {
        p[0][r] += vdn * fast_tanh(acc0[r]);
        p[1][r] += vdn * fast_tanh(acc1[r]);
      }
    }

    // reduce over N (the 16 lanes of each half); C/D: M = r + 8*(lane>>4)
    #pragma unroll
    for (int mt = 0; mt < 2; ++mt) {
      #pragma unroll
      for (int r = 0; r < 8; ++r) {
        float v = p[mt][r];
        v += __shfl_xor(v, 1, 16);
        v += __shfl_xor(v, 2, 16);
        v += __shfl_xor(v, 4, 16);
        v += __shfl_xor(v, 8, 16);
        p[mt][r] = v;
      }
    }
    if (l15 == 0) {
      #pragma unroll
      for (int mt = 0; mt < 2; ++mt)
        #pragma unroll
        for (int r = 0; r < 8; ++r)
          s_lds[mrow0 + mt * 16 + lhalf * 8 + r] = p[mt][r];
    }
    __syncthreads();

    // ---------------- softmax over tx (256 scores, 256 threads) ----------------
    const float sv = s_lds[tid];
    float m = sv;
    m = fmaxf(m, __shfl_xor(m, 16, 32));
    m = fmaxf(m, __shfl_xor(m,  8, 32));
    m = fmaxf(m, __shfl_xor(m,  4, 32));
    m = fmaxf(m, __shfl_xor(m,  2, 32));
    m = fmaxf(m, __shfl_xor(m,  1, 32));
    if (lane == 0) red_s[wave] = m;
    __syncthreads();
    float mm = red_s[0];
    #pragma unroll
    for (int w = 1; w < 8; ++w) mm = fmaxf(mm, red_s[w]);
    __syncthreads();  // red_s reads done before re-use

    const float e = __expf(sv - mm);
    s_lds[tid] = e;
    float ssum = e;
    ssum += __shfl_xor(ssum, 16, 32);
    ssum += __shfl_xor(ssum,  8, 32);
    ssum += __shfl_xor(ssum,  4, 32);
    ssum += __shfl_xor(ssum,  2, 32);
    ssum += __shfl_xor(ssum,  1, 32);
    if (lane == 0) red_s[wave] = ssum;
    __syncthreads();
    float tot = 0.f;
    #pragma unroll
    for (int w = 0; w < 8; ++w) tot += red_s[w];
    const float inv = 1.0f / tot;

    // ---------------- qtd[ty][c] = inv * sum_tx e[tx] * X[tx][c] ----------------
    float acc = 0.f;
    #pragma unroll 8
    for (int tx = 0; tx < TXN; ++tx) {
      acc += s_lds[tx] * (float)Xs[tx * LDX + tid];
    }
    out[((size_t)b * TYN + ty) * CC + tid] = acc * inv;
  }
}

extern "C" void kernel_launch(void* const* d_in, const int* in_sizes, int n_in,
                              void* d_out, int out_size, void* d_ws, size_t ws_size,
                              hipStream_t stream) {
  const float* x  = (const float*)d_in[0];
  const float* y  = (const float*)d_in[1];
  const float* Wd = (const float*)d_in[2];
  const float* vd = (const float*)d_in[3];
  float* out = (float*)d_out;

  dim3 grid(BB * (TYN / TG));   // 512 workgroups
  dim3 block(256);              // 8 wave32s
  dot_attention_kernel<<<grid, block, 0, stream>>>(x, y, Wd, vd, out);
}
